// ArcFace_1133871366196
// MI455X (gfx1250) — matrix operand order repaired and checked by reading, not compiled, not run on hardware
//
#include <hip/hip_runtime.h>

#define B_ROWS 512
#define D_K    512
#define NT     64
#define KSTEP  32
#define NSTEP  (D_K / KSTEP)
#define LROW   40   // LDS row stride in bf16 halves (80B) -> conflict-free 16B loads

typedef __attribute__((ext_vector_type(16))) __bf16 v16bf;
typedef __attribute__((ext_vector_type(8)))  __bf16 v8bf;
typedef __attribute__((ext_vector_type(8)))  float  v8f;

#if defined(__HIP_DEVICE_COMPILE__)
#if __has_builtin(__builtin_amdgcn_global_load_async_to_lds_b128)
#define HAVE_ASYNC_LDS 1
#endif
#endif

// Exact parameter types for the async builtin:
//   param0: v4i32 in AS(1) (global, printed as __device__), param1: v4i32 in AS(3) (LDS)
typedef int v4i_ __attribute__((vector_size(16)));
typedef __attribute__((address_space(1))) v4i_* p1_v4i;
typedef __attribute__((address_space(3))) v4i_* p3_v4i;

__device__ __forceinline__ void async_copy_b128(const void* gsrc, const void* ldst) {
#ifdef HAVE_ASYNC_LDS
    __builtin_amdgcn_global_load_async_to_lds_b128(
        (p1_v4i)(unsigned long long)gsrc,
        (p3_v4i)(unsigned)(unsigned long long)ldst, 0, 0);
#endif
}

__device__ __forceinline__ void wait_async_zero() {
#ifdef HAVE_ASYNC_LDS
#if __has_builtin(__builtin_amdgcn_s_wait_asynccnt)
    __builtin_amdgcn_s_wait_asynccnt(0);
#else
    asm volatile("s_wait_asynccnt 0x0" ::: "memory");
#endif
#endif
}

__device__ __forceinline__ unsigned int f2bf(float f) {
    unsigned int u = __float_as_uint(f);
    u += 0x7FFFu + ((u >> 16) & 1u);       // round-to-nearest-even
    return u >> 16;
}

__device__ __forceinline__ float wave_reduce_sum(float v) {
#pragma unroll
    for (int off = 16; off > 0; off >>= 1) v += __shfl_xor(v, off, 32);
    return v;
}

// --- Pass 1a: weight row inverse norms (one wave32 per row) ---------------
__global__ __launch_bounds__(256) void nw_kernel(const float* __restrict__ W,
                                                 float* __restrict__ wscale, int C) {
    int row = blockIdx.x * 8 + (threadIdx.x >> 5);
    if (row >= C) return;
    int lane = threadIdx.x & 31;
    const float4* p = (const float4*)(W + (size_t)row * D_K);
    float s = 0.f;
#pragma unroll
    for (int i = 0; i < 4; ++i) {
        float4 v = p[lane + i * 32];
        s += v.x * v.x + v.y * v.y + v.z * v.z + v.w * v.w;
    }
    s = wave_reduce_sum(s);
    if (lane == 0) wscale[row] = 1.0f / fmaxf(sqrtf(s), 1e-12f);
}

// --- Pass 1b: normalize input rows, emit bf16 A into ws -------------------
__global__ __launch_bounds__(256) void ni_kernel(const float* __restrict__ X,
                                                 unsigned short* __restrict__ Abf) {
    int row  = blockIdx.x * 8 + (threadIdx.x >> 5);   // 64 blocks -> 512 rows
    int lane = threadIdx.x & 31;
    const float4* p = (const float4*)(X + (size_t)row * D_K);
    float4 v[4];
    float  s = 0.f;
#pragma unroll
    for (int i = 0; i < 4; ++i) {
        v[i] = p[lane + i * 32];
        s += v[i].x * v[i].x + v[i].y * v[i].y + v[i].z * v[i].z + v[i].w * v[i].w;
    }
    s = wave_reduce_sum(s);
    float sc = 1.0f / fmaxf(sqrtf(s), 1e-12f);
#pragma unroll
    for (int i = 0; i < 4; ++i) {
        uint2 pk;
        pk.x = f2bf(v[i].x * sc) | (f2bf(v[i].y * sc) << 16);
        pk.y = f2bf(v[i].z * sc) | (f2bf(v[i].w * sc) << 16);
        *(uint2*)(Abf + (size_t)row * D_K + (lane + i * 32) * 4) = pk;
    }
}

// --- Pass 2: fused bf16-WMMA GEMM + ArcFace margin epilogue ---------------
// Double-buffered LDS, software-pipelined global fetch (async-to-LDS for A).
__global__ __launch_bounds__(512) void arcface_kernel(
    const unsigned short* __restrict__ Abf,   // [512][512] normalized bf16
    const float*          __restrict__ W,     // [C][512] fp32
    const float*          __restrict__ wscale,// [C]
    const int*            __restrict__ label, // [512]
    float*                __restrict__ out, int C)
{
    __shared__ unsigned short Alds[2][B_ROWS * LROW]; // 80 KB
    __shared__ unsigned short Blds[2][NT * LROW];     // 10 KB
    __shared__ int            lab[B_ROWS];            //  2 KB

    const int tid = threadIdx.x;
    const int n0  = blockIdx.x * NT;

    lab[tid] = label[tid];

    // B-tile loader assignment: thread -> (weight row, float4 chunk)
    const int ln = tid >> 3;                 // 0..63
    const int lc = tid & 7;                  // 0..7
    int wrow = n0 + ln; if (wrow >= C) wrow = C - 1;
    const float  wsc   = wscale[wrow];
    const float* wbase = W + (size_t)wrow * D_K + lc * 4;

    // wave / tile mapping: 16 waves -> 8 M-groups x 2 N-groups
    const int wave = tid >> 5;
    const int lane = tid & 31;
    const int wn   = wave & 1;               // N cols [wn*32, wn*32+32)
    const int wm   = wave >> 1;              // M rows [wm*64, wm*64+64)
    const int half = lane >> 4;
    const int l16  = lane & 15;

    v8f acc[4][2] = {};

    const unsigned short* arow       = Abf + (size_t)tid * D_K;
    unsigned short*       aldsrow[2] = { &Alds[0][tid * LROW], &Alds[1][tid * LROW] };

    // ---- prologue: stage step 0 ----
#ifdef HAVE_ASYNC_LDS
#pragma unroll
    for (int c = 0; c < 4; ++c) async_copy_b128(arow + c * 8, aldsrow[0] + c * 8);
#else
    uint4 areg[4];
#pragma unroll
    for (int c = 0; c < 4; ++c) areg[c] = *(const uint4*)(arow + c * 8);
#endif
    float4 wv = *(const float4*)(wbase);

    for (int s = 0; s < NSTEP; ++s) {
        const int buf = s & 1;
        // stage B for this step (fuses norm scale + fp32->bf16 cvt)
        uint2 pk;
        pk.x = f2bf(wv.x * wsc) | (f2bf(wv.y * wsc) << 16);
        pk.y = f2bf(wv.z * wsc) | (f2bf(wv.w * wsc) << 16);
        *(uint2*)(&Blds[buf][ln * LROW + lc * 4]) = pk;
#ifndef HAVE_ASYNC_LDS
        // stage A regs -> LDS for this step
#pragma unroll
        for (int c = 0; c < 4; ++c)
            *(uint4*)(&Alds[buf][tid * LROW + c * 8]) = areg[c];
#endif
        wait_async_zero();          // A async copies for this buffer done
        __syncthreads();

        // ---- issue next step's global fetches (latency overlaps WMMAs) ----
        if (s + 1 < NSTEP) {
            const int knext = (s + 1) * KSTEP;
#ifdef HAVE_ASYNC_LDS
#pragma unroll
            for (int c = 0; c < 4; ++c)
                async_copy_b128(arow + knext + c * 8, aldsrow[(s + 1) & 1] + c * 8);
#else
#pragma unroll
            for (int c = 0; c < 4; ++c)
                areg[c] = *(const uint4*)(arow + knext + c * 8);
#endif
            wv = *(const float4*)(wbase + knext);
            if (s + 2 < NSTEP) __builtin_prefetch(wbase + knext + KSTEP, 0, 0);
        }

        // ---- WMMAs on current buffer ----
        // B fragments: lane l16 = column, half interleaves K {0..7,16..23}/{8..15,24..31}
        v16bf bfrag[2];
#pragma unroll
        for (int j = 0; j < 2; ++j) {
            const __bf16* bp = (const __bf16*)&Blds[buf][(wn * 32 + j * 16 + l16) * LROW + half * 8];
            v8bf lo = *(const v8bf*)bp;
            v8bf hi = *(const v8bf*)(bp + 16);
            bfrag[j] = __builtin_shufflevector(lo, hi, 0,1,2,3,4,5,6,7,8,9,10,11,12,13,14,15);
        }
        // A fragments: lane l16 = row, half interleaves K {0..7,16..23}/{8..15,24..31}
#pragma unroll
        for (int i = 0; i < 4; ++i) {
            const __bf16* ap = (const __bf16*)&Alds[buf][(wm * 64 + i * 16 + l16) * LROW + half * 8];
            v8bf lo = *(const v8bf*)ap;
            v8bf hi = *(const v8bf*)(ap + 16);
            v16bf afrag = __builtin_shufflevector(lo, hi, 0,1,2,3,4,5,6,7,8,9,10,11,12,13,14,15);
#pragma unroll
            for (int j = 0; j < 2; ++j)
                acc[i][j] = __builtin_amdgcn_wmma_f32_16x16x32_bf16(
                    false, afrag, false, bfrag[j], (short)0, acc[i][j], false, false);
        }
    }

    // ArcFace margin epilogue (fused, register-resident)
    const float Sc   = 30.0f;
    const float COSM = 0.8775825618903728f;   // cos(0.5)
    const float SINM = 0.4794255386042030f;   // sin(0.5)
    const float THR  = -0.8775825618903728f;  // cos(pi-0.5)
    const float MMc  = 0.2397127693021015f;   // sin(pi-0.5)*0.5

#pragma unroll
    for (int i = 0; i < 4; ++i) {
        const int mbase = wm * 64 + i * 16 + half * 8;
#pragma unroll
        for (int j = 0; j < 2; ++j) {
            const int ncol = n0 + wn * 32 + j * 16 + l16;
            if (ncol < C) {
#pragma unroll
                for (int r = 0; r < 8; ++r) {
                    const int m = mbase + r;
                    float c   = fminf(fmaxf(acc[i][j][r], -1.f), 1.f);
                    float s2  = fminf(fmaxf(1.f - c * c, 1e-9f), 1.f);
                    float phi = c * COSM - sqrtf(s2) * SINM;
                    phi = (c > THR) ? phi : (c - MMc);
                    float v = (lab[m] == ncol) ? phi : c;
                    out[(size_t)m * C + ncol] = v * Sc;
                }
            }
        }
    }
}

extern "C" void kernel_launch(void* const* d_in, const int* in_sizes, int n_in,
                              void* d_out, int out_size, void* d_ws, size_t ws_size,
                              hipStream_t stream) {
    const float* inp    = (const float*)d_in[0];
    const int*   label  = (const int*)d_in[1];
    const float* weight = (const float*)d_in[2];
    float*       out    = (float*)d_out;
    const int C = 100000;

    // ws layout: [0, 400128) wscale (C floats, padded to 16B), then bf16 A (512KB)
    float*          wscale = (float*)d_ws;
    unsigned short* Abf    = (unsigned short*)((char*)d_ws + 400128);

    nw_kernel<<<(C + 7) / 8, 256, 0, stream>>>(weight, wscale, C);
    ni_kernel<<<B_ROWS / 8, 256, 0, stream>>>(inp, Abf);
    arcface_kernel<<<(C + NT - 1) / NT, 512, 0, stream>>>(Abf, weight, wscale, label, out, C);
}
